// ExtGNOBNO_7103875907956
// MI455X (gfx1250) — matrix operand order repaired and checked by reading, not compiled, not run on hardware
//
#include <hip/hip_runtime.h>

typedef __attribute__((ext_vector_type(16))) _Float16 v16h;
typedef __attribute__((ext_vector_type(8)))  _Float16 v8h;
typedef __attribute__((ext_vector_type(8)))  float    v8f;

constexpr int kB   = 4;
constexpr int kN   = 16384;
constexpr int kE   = 262144;
constexpr int kC   = 64;
constexpr int kFC  = 128;
constexpr int EPB  = 128;     // edges per tile (8 waves x 16-edge WMMA tiles)
constexpr int ETILES = 4;     // edge tiles per block (amortize w2 staging)
constexpr int NTHREADS = 256; // 8 wave32

__global__ __launch_bounds__(NTHREADS)
void gno_block_wmma(const float* __restrict__ nodes_x,
                    const float* __restrict__ nodes_y,
                    const float* __restrict__ x,
                    const int*   __restrict__ edges,     // (B,E,2) flattened
                    const float* __restrict__ weights,   // (B,E)
                    const float* __restrict__ w1,        // (4,128)
                    const float* __restrict__ b1,        // (128)
                    const float* __restrict__ w2,        // (128,64)
                    const float* __restrict__ b2,        // (64)
                    float* __restrict__ out)             // (B,C,N), pre-zeroed
{
    __shared__ __align__(32) _Float16 h_lds[EPB][kFC]; // 32 KB : A operands (edges x FC)
    __shared__ __align__(32) _Float16 w2t[kC][kFC];    // 16 KB : B operands, transposed
    __shared__ float coords[EPB][4];
    __shared__ float ew[EPB];
    __shared__ int   esrc[EPB];
    __shared__ int   etgt[EPB];
    __shared__ float b2s[kC];

    const int tid = threadIdx.x;
    constexpr int blocksPerBatch = kE / (EPB * ETILES);   // 512
    const int b    = blockIdx.x / blocksPerBatch;
    const int eblk = (blockIdx.x % blocksPerBatch) * (EPB * ETILES);

    // ---- Phase 0 (once per block): stage w2^T (f16) and b2 into LDS ----
    for (int i = tid; i < kFC * kC; i += NTHREADS) {
        int k = i >> 6;        // 0..127
        int n = i & (kC - 1);  // 0..63
        w2t[n][k] = (_Float16)w2[k * kC + n];
    }
    if (tid < kC) b2s[tid] = b2[tid];

    const int wave  = tid >> 5;       // 0..7 : one 16-edge tile per wave
    const int lane  = tid & 31;
    const int col16 = lane & 15;      // column within 16-wide N tile / A row
    const int hi    = lane >> 4;      // half-wave selector
    const int m0    = wave * 16;      // first edge of this wave's tile

    const size_t xb = (size_t)b * kC * kN;
    const float* __restrict__ xbase = x + xb;    // uniform base -> SADDR form
    float* __restrict__       obase = out + xb;  // uniform base -> SADDR form

    for (int it = 0; it < ETILES; ++it) {
        const int e0 = eblk + it * EPB;

        // ---- Phase 1: gather per-edge indices, weights, coordinates ----
        if (tid < EPB) {
            int e = e0 + tid;
            long long ebase = ((long long)b * kE + e) * 2;
            int s = edges[ebase + 0];
            int t = edges[ebase + 1];
            esrc[tid] = s;
            etgt[tid] = t;
            ew[tid]   = weights[(long long)b * kE + e];
            const float* px = &nodes_x[((long long)b * kN + s) * 2];
            const float* py = &nodes_y[((long long)b * kN + t) * 2];
            coords[tid][0] = px[0];
            coords[tid][1] = px[1];
            coords[tid][2] = py[0];
            coords[tid][3] = py[1];
        }
        __syncthreads();

        // ---- Phase 2: h = gelu_exact(coords @ w1 + b1) -> f16 in LDS ----
        for (int i = tid; i < EPB * kFC; i += NTHREADS) {
            int el = i >> 7;          // edge in tile
            int f  = i & (kFC - 1);   // hidden channel
            float z = coords[el][0] * w1[0 * kFC + f]
                    + coords[el][1] * w1[1 * kFC + f]
                    + coords[el][2] * w1[2 * kFC + f]
                    + coords[el][3] * w1[3 * kFC + f]
                    + b1[f];
            float g = 0.5f * z * (1.0f + erff(z * 0.70710678118654752f));
            h_lds[el][f] = (_Float16)g;
        }
        __syncthreads();

        // ---- Phase 3: ker = h @ w2 + b2 via V_WMMA_F32_16X16X32_F16 ----
        // A fragments for the 4 K-tiles (K=128 total, 32 per WMMA).
        // Lane l holds row (l&15); halves K = k0 + hi*8 + {0..7} and +16.
        v16h a[4];
#pragma unroll
        for (int kk = 0; kk < 4; ++kk) {
            const _Float16* pa = &h_lds[m0 + col16][kk * 32 + hi * 8];
            v8h lo  = *(const v8h*)(pa);
            v8h hi8 = *(const v8h*)(pa + 16);
#pragma unroll
            for (int j = 0; j < 8; ++j) { a[kk][j] = lo[j]; a[kk][8 + j] = hi8[j]; }
        }

        // Per-lane row metadata, hoisted out of the N-tile loop (register resident).
        int   s8[8], t8[8];
        float w8[8];
#pragma unroll
        for (int r = 0; r < 8; ++r) {
            const int el = m0 + hi * 8 + r;
            s8[r] = esrc[el];
            t8[r] = etgt[el];
            w8[r] = ew[el];
        }

#pragma unroll
        for (int nt = 0; nt < 4; ++nt) {        // 4 N-tiles of 16 channels
            const int col = nt * 16 + col16;

            // C init with bias broadcast along the column.
            v8f c;
            const float bb = b2s[col];
#pragma unroll
            for (int r = 0; r < 8; ++r) c[r] = bb;

#pragma unroll
            for (int kk = 0; kk < 4; ++kk) {
                // B fragment: lane holds column (lane&15),
                // K = kk*32 + hi*8 + {0..7} and +16, matching A.
                const _Float16* pb = &w2t[col][kk * 32 + hi * 8];
                v8h blo = *(const v8h*)(pb);
                v8h bhi = *(const v8h*)(pb + 16);
                v16h bv;
#pragma unroll
                for (int j = 0; j < 8; ++j) { bv[j] = blo[j]; bv[8 + j] = bhi[j]; }
                c = __builtin_amdgcn_wmma_f32_16x16x32_f16(
                        false, a[kk], false, bv, (short)0, c, false, false);
            }

            // Gather all 8 xs first (8 loads in flight), then scatter atomics.
            const unsigned co = (unsigned)col * kN;
            float xsv[8];
#pragma unroll
            for (int r = 0; r < 8; ++r)
                xsv[r] = xbase[co + (unsigned)s8[r]];
#pragma unroll
            for (int r = 0; r < 8; ++r)
                atomicAdd(&obase[co + (unsigned)t8[r]], w8[r] * c[r] * xsv[r]);
        }
        __syncthreads();   // protect LDS before next tile's phase 1/2 writes
    }
}

extern "C" void kernel_launch(void* const* d_in, const int* in_sizes, int n_in,
                              void* d_out, int out_size, void* d_ws, size_t ws_size,
                              hipStream_t stream) {
    const float* nodes_x = (const float*)d_in[0];
    const float* nodes_y = (const float*)d_in[1];
    const float* x       = (const float*)d_in[2];
    const int*   edges   = (const int*)  d_in[3];
    const float* weights = (const float*)d_in[4];
    const float* w1      = (const float*)d_in[5];
    const float* b1      = (const float*)d_in[6];
    const float* w2      = (const float*)d_in[7];
    const float* b2      = (const float*)d_in[8];
    float* out = (float*)d_out;

    // Output accumulated via atomics -> must start from zero every call.
    hipMemsetAsync(out, 0, (size_t)kB * kC * kN * sizeof(float), stream);

    dim3 grid(kB * (kE / (EPB * ETILES)));   // 4 * 512 = 2048 blocks
    dim3 block(NTHREADS);
    gno_block_wmma<<<grid, block, 0, stream>>>(nodes_x, nodes_y, x, edges, weights,
                                               w1, b1, w2, b2, out);
}
